// RNN_77051713290272
// MI455X (gfx1250) — matrix operand order repaired and checked by reading, not compiled, not run on hardware
//
#include <hip/hip_runtime.h>
#include <math.h>

// Persistent single-wave RNN scan for MI455X (gfx1250, wave32).
// h_{t+1} = tanh(x_t * w_in + W_hh h_t + bias), T sequential steps.
// W_hh lives in VGPRs as split-f16 (hi+lo) in WMMA-B layout with a permuted
// K ordering (pi(2l)=l, pi(2l+1)=l+32) so each lane publishes its two h
// values with a single packed ds_store_b32. Each step: 16 x
// v_wmma_f32_16x16x32_f16 as 4 depth-4 accumulation chains issued
// breadth-first (all accumulators live -> distinct register sets -> all
// WMMAs stream back-to-back and the D-read hazard windows overlap) for the
// 64x64 padded matvec at ~f32 accuracy.

#define H 50
#define HP 64

typedef __attribute__((ext_vector_type(16))) _Float16 v16h;
typedef __attribute__((ext_vector_type(8)))  _Float16 v8h;
typedef __attribute__((ext_vector_type(2)))  _Float16 v2h;
typedef __attribute__((ext_vector_type(8)))  float    v8f;

// K-permutation: position p in the contraction holds h[pi(p)].
static __device__ __forceinline__ int kperm(int p) {
  return (p & 1) ? (p >> 1) + 32 : (p >> 1);
}

static __device__ __forceinline__ float fast_tanh(float x) {
#if __has_builtin(__builtin_amdgcn_tanhf)
  return __builtin_amdgcn_tanhf(x);            // v_tanh_f32 on gfx1250
#elif __has_builtin(__builtin_amdgcn_tanh_f32)
  return __builtin_amdgcn_tanh_f32(x);
#else
  return tanhf(x);
#endif
}

static __device__ __forceinline__ void wait_lds() {
#if __has_builtin(__builtin_amdgcn_s_wait_dscnt)
  __builtin_amdgcn_s_wait_dscnt(0);
#else
  asm volatile("s_wait_dscnt 0" ::: "memory");
#endif
}

__global__ void __launch_bounds__(32, 1)
rnn_scan_wmma(const float* __restrict__ seq, const float* __restrict__ W_ih,
              const float* __restrict__ W_hh, const float* __restrict__ b_ih,
              const float* __restrict__ b_hh, const float* __restrict__ W_out,
              const float* __restrict__ b_out, const float* __restrict__ h0,
              float* __restrict__ out, int T) {
  __shared__ __align__(16) _Float16 h16[HP];  // h in permuted K order (f16)
  __shared__ float hfin[HP];                  // final h for output reduce

  const int l    = threadIdx.x;   // 0..31, one full wave (EXEC all ones)
  const int n    = l & 15;        // column within a 16-wide output chunk
  const int half = l >> 4;        // lane half selects K sub-range in A/B

  // Per-chunk bias and input weight: chunk c covers output rows 16c + n.
  float biasr[4], winr[4];
#pragma unroll
  for (int c = 0; c < 4; ++c) {
    int row = 16 * c + n;
    bool ok = row < H;
    biasr[c] = ok ? (b_ih[row] + b_hh[row]) : 0.0f;
    winr[c]  = ok ? W_ih[row] : 0.0f;
  }

  // Preload W_hh^T blocks as split-f16 (hi + lo) in WMMA B (32x16) layout,
  // with the K dimension permuted by kperm so it matches the packed h store:
  // position p holds W_hh[row, kperm(p)]; lane group `half` holds positions
  // 16*half + e (e = v16h element). Zero-padded past H.
  v16h Bhi[4][2], Blo[4][2];
#pragma unroll
  for (int c = 0; c < 4; ++c) {
    int row = 16 * c + n;
#pragma unroll
    for (int kc = 0; kc < 2; ++kc) {
#pragma unroll
      for (int e = 0; e < 16; ++e) {
        int p = 32 * kc + 16 * half + e;   // position in contraction
        int k = kperm(p);                  // actual W_hh column
        float w = (row < H && k < H) ? W_hh[row * H + k] : 0.0f;
        _Float16 whi = (_Float16)w;
        _Float16 wlo = (_Float16)(w - (float)whi);
        Bhi[c][kc][e] = whi;
        Blo[c][kc][e] = wlo;
      }
    }
  }

  // Lane l owns h[l] and h[l+32] (zero-padded past H); published at packed
  // positions 2l, 2l+1.
  float h_a = (l < H) ? h0[l] : 0.0f;
  float h_b = (l + 32 < H) ? h0[l + 32] : 0.0f;

  const int a_off = half * 16;  // A-layout K offset for this lane group

  // One recurrence step; updates h_a/h_b from scalar input x.
  auto step = [&](float x) {
    // Publish both h values with one packed b32 store. The per-wave LDS
    // pipe is in-order, so the loads below observe this store; the compiler
    // inserts the data-dependency s_wait_dscnt before WMMA use.
    v2h hp;
    hp[0] = (_Float16)h_a;
    hp[1] = (_Float16)h_b;
    *(v2h*)&h16[2 * l] = hp;

    // Build A (h replicated across M rows) in contiguous K order (element e holds position 16*half + e), matching B.
    v8h a0lo = *(const v8h*)(&h16[a_off]);        // positions 0-7   / 16-23
    v8h a0hi = *(const v8h*)(&h16[8 + a_off]);    // positions 8-15  / 24-31
    v8h a1lo = *(const v8h*)(&h16[32 + a_off]);   // positions 32-39 / 48-55
    v8h a1hi = *(const v8h*)(&h16[40 + a_off]);   // positions 40-47 / 56-63
    v16h A0, A1;
#pragma unroll
    for (int e = 0; e < 8; ++e) {
      A0[e] = a0lo[e]; A0[8 + e] = a0hi[e];
      A1[e] = a1lo[e]; A1[8 + e] = a1hi[e];
    }

    // x*w_in + bias per chunk (overlaps with LDS latency / WMMA issue).
    float cb[4];
#pragma unroll
    for (int c = 0; c < 4; ++c) cb[c] = fmaf(x, winr[c], biasr[c]);

    // 4 depth-4 accumulation chains, issued breadth-first so all 4
    // accumulators are live simultaneously: the allocator must give them
    // distinct register sets, all 16 WMMAs can stream back-to-back, and
    // the per-chain D-read hazard windows overlap instead of stacking.
    // A0-sourced rounds come first (their LDS loads complete earliest).
    v8f acc[4];
    {
      v8f z = {};
#pragma unroll
      for (int c = 0; c < 4; ++c)       // round 1: A0 x Bhi (seed inline 0)
        acc[c] = __builtin_amdgcn_wmma_f32_16x16x32_f16(false, A0, false,
                     Bhi[c][0], (short)0, z, false, false);
#pragma unroll
      for (int c = 0; c < 4; ++c)       // round 2: A0 x Blo
        acc[c] = __builtin_amdgcn_wmma_f32_16x16x32_f16(false, A0, false,
                     Blo[c][0], (short)0, acc[c], false, false);
#pragma unroll
      for (int c = 0; c < 4; ++c)       // round 3: A1 x Bhi
        acc[c] = __builtin_amdgcn_wmma_f32_16x16x32_f16(false, A1, false,
                     Bhi[c][1], (short)0, acc[c], false, false);
#pragma unroll
      for (int c = 0; c < 4; ++c)       // round 4: A1 x Blo
        acc[c] = __builtin_amdgcn_wmma_f32_16x16x32_f16(false, A1, false,
                     Blo[c][1], (short)0, acc[c], false, false);
    }

    // Every D row is identical; element 0 is y[16c + n] for this lane.
    float y[4];
#pragma unroll
    for (int c = 0; c < 4; ++c) y[c] = acc[c][0] + cb[c];

    bool lo16 = l < 16;
    h_a = fast_tanh(lo16 ? y[0] : y[1]);  // y[l]
    h_b = fast_tanh(lo16 ? y[2] : y[3]);  // y[l+32]
  };

  // 4-deep rotating prefetch of the (wave-uniform) scalar inputs.
  float xs[4];
#pragma unroll
  for (int j = 0; j < 4; ++j) xs[j] = seq[(j < T) ? j : (T - 1)];

  int t = 0;
  // Main loop: t + 8 <= T guarantees all prefetch indices are in-bounds,
  // so no clamping SALU in the hot loop.
  for (; t + 8 <= T; t += 4) {
#pragma unroll
    for (int j = 0; j < 4; ++j) {
      float x = xs[j];
      xs[j] = seq[t + 4 + j];  // prefetch 4 steps ahead, always in-bounds
      step(x);
    }
  }
  for (; t < T; ++t) step(seq[t]);  // tail (<= 7 steps, loads from cache)

  // out = h_last . W_out + b_out  (one-time scalar reduce)
  hfin[l]      = h_a;
  hfin[l + 32] = h_b;
  wait_lds();
  if (l == 0) {
    float s = b_out[0];
    for (int i = 0; i < H; ++i) s = fmaf(hfin[i], W_out[i], s);
    out[0] = s;
  }
}

extern "C" void kernel_launch(void* const* d_in, const int* in_sizes, int n_in,
                              void* d_out, int out_size, void* d_ws, size_t ws_size,
                              hipStream_t stream) {
  (void)n_in; (void)out_size; (void)d_ws; (void)ws_size;
  const float* seq   = (const float*)d_in[0];
  const float* W_ih  = (const float*)d_in[1];
  const float* W_hh  = (const float*)d_in[2];
  const float* b_ih  = (const float*)d_in[3];
  const float* b_hh  = (const float*)d_in[4];
  const float* W_out = (const float*)d_in[5];
  const float* b_out = (const float*)d_in[6];
  const float* h0    = (const float*)d_in[7];
  const int T = in_sizes[0];

  // Inherently sequential scan: one persistent wave32 block, WMMA needs a
  // full wave (EXEC all ones), so blockDim = 32, grid = 1.
  rnn_scan_wmma<<<1, 32, 0, stream>>>(seq, W_ih, W_hh, b_ih, b_hh,
                                      W_out, b_out, h0, (float*)d_out, T);
}
